// DNDecoder_10007273800198
// MI455X (gfx1250) — matrix operand order repaired and checked by reading, not compiled, not run on hardware
//
#include <hip/hip_runtime.h>
#include <hip/hip_bf16.h>

typedef __attribute__((ext_vector_type(16))) __bf16 v16bf;
typedef __attribute__((ext_vector_type(8)))  __bf16 v8bf;
typedef __attribute__((ext_vector_type(4)))  __bf16 v4bf;
typedef __attribute__((ext_vector_type(8)))  float  v8f;
typedef __attribute__((ext_vector_type(4)))  float  v4f;

#define H   128
#define H2  256
#define NEG_SLOPE 0.01f

// ---------------------------------------------------------------------------
// Stage 0a: pack W1 (f32, [128 x 256] row-major, W1[o*256+c]) into transposed
// bf16 Wp[j*128+h] where j<128 -> W1[j,h] (A half), j>=128 -> W1[j-128,128+h].
// Each B-fragment load in the GEMM becomes a contiguous 8-element run.
// ---------------------------------------------------------------------------
__global__ void pack_w_kernel(const float* __restrict__ w1,
                              __bf16* __restrict__ wp) {
    int j = blockIdx.x;     // 0..255 output column of the fused GEMM
    int h = threadIdx.x;    // 0..127 reduction index
    float v = (j < H) ? w1[j * H2 + h] : w1[(j - H) * H2 + H + h];
    wp[j * H + h] = (__bf16)v;
}

// ---------------------------------------------------------------------------
// Stage 0b: convert z (f32) to bf16, zero-padding up to a multiple-of-16 rows.
// ---------------------------------------------------------------------------
__global__ void pack_z_kernel(const float* __restrict__ z,
                              __bf16* __restrict__ zb,
                              int total_real, int total_pad) {
    int idx = (blockIdx.x * blockDim.x + threadIdx.x) * 4;
    if (idx >= total_pad) return;
    v4bf o;
    if (idx + 3 < total_real) {
        v4f in = *(const v4f*)(z + idx);
        o[0] = (__bf16)in[0]; o[1] = (__bf16)in[1];
        o[2] = (__bf16)in[2]; o[3] = (__bf16)in[3];
    } else {
        #pragma unroll
        for (int k = 0; k < 4; ++k) {
            float v = (idx + k < total_real) ? z[idx + k] : 0.0f;
            o[k] = (__bf16)v;
        }
    }
    *(v4bf*)(zb + idx) = o;
}

// ---------------------------------------------------------------------------
// Stage 1: AB[n_pad x 256] = zb[n_pad x 128] @ Wp^T via WMMA bf16.
// One wave computes one 16(M) x 16(N) f32 tile, K = 128 = 4 x 32.
// Block = 256 threads = 8 waves; grid = (n_pad/16, 2) covering 16 N-tiles.
//
// A-fragment (16x32 bf16): lane l holds row M=l&15; kb=(l>>4)*8;
//   elements 0..7 = K[kb..kb+7], elements 8..15 = K[kb+16..kb+23].
// B-fragment (32x16 bf16): lane l holds col N=l&15; kb=(l>>4)*8;
//   elements 0..7 = K[kb..kb+7], elements 8..15 = K[kb+16..kb+23].
// C/D (16x16 f32): lane l holds col N=l&15; VGPR r holds row M=(l>>4)*8+r.
// ---------------------------------------------------------------------------
__global__ void gemm_node_kernel(const __bf16* __restrict__ zb,
                                 const __bf16* __restrict__ wp,
                                 float* __restrict__ ab) {
    const int lane  = threadIdx.x & 31;
    const int wave  = threadIdx.x >> 5;
    const int mtile = blockIdx.x;                 // 16 nodes
    const int ntile = blockIdx.y * 8 + wave;      // 16 fused-output columns
    const int half  = lane >> 4;                  // 0 or 1
    const int l15   = lane & 15;

    const __bf16* za = zb + (size_t)(mtile * 16 + l15) * H + half * 8;
    const __bf16* wb = wp + (size_t)(ntile * 16 + l15) * H + half * 8;

    v8f c = {0.f, 0.f, 0.f, 0.f, 0.f, 0.f, 0.f, 0.f};
    #pragma unroll
    for (int kt = 0; kt < 4; ++kt) {
        const int k0 = kt * 32;
        v8bf a_lo = *(const v8bf*)(za + k0);          // K = kb   .. kb+7
        v8bf a_hi = *(const v8bf*)(za + k0 + 16);     // K = kb+16.. kb+23
        v16bf a = __builtin_shufflevector(a_lo, a_hi,
                   0,1,2,3,4,5,6,7,8,9,10,11,12,13,14,15);
        v8bf b_lo = *(const v8bf*)(wb + k0);          // K = kb   .. kb+7
        v8bf b_hi = *(const v8bf*)(wb + k0 + 16);     // K = kb+16.. kb+23
        v16bf b = __builtin_shufflevector(b_lo, b_hi,
                   0,1,2,3,4,5,6,7,8,9,10,11,12,13,14,15);
        // 8 args: (neg_a, A, neg_b, B, c_mod, C, reuse_a, reuse_b)
        c = __builtin_amdgcn_wmma_f32_16x16x32_bf16(
                false, a, false, b, (short)0, c, false, false);
    }

    float* out = ab + (size_t)(mtile * 16 + half * 8) * H2 + ntile * 16 + l15;
    #pragma unroll
    for (int r = 0; r < 8; ++r)
        out[(size_t)r * H2] = c[r];
}

// ---------------------------------------------------------------------------
// Stage 2: per-edge combine. One wave per edge (x8 edges per wave to amortize
// the b1/W2 register preload). Lane handles 4 channels o = lane*4..lane*4+3.
// AB (102 MB) lives in the 192 MB L2 -> this runs at L2 gather bandwidth.
// ---------------------------------------------------------------------------
__global__ void edge_kernel(const float* __restrict__ ab,
                            const int*  __restrict__ eidx,
                            const float* __restrict__ b1,
                            const float* __restrict__ w2,
                            const float* __restrict__ b2,
                            float* __restrict__ out,
                            int n_edges) {
    const int lane = threadIdx.x & 31;
    const int wave = threadIdx.x >> 5;
    const int o    = lane * 4;

    const v4f vb1 = *(const v4f*)(b1 + o);
    const v4f vw2 = *(const v4f*)(w2 + o);
    const float bias2 = b2[0];

    const int EPW   = 8;
    const int ebase = (blockIdx.x * (blockDim.x >> 5) + wave) * EPW;

    for (int i = 0; i < EPW; ++i) {
        const int e = ebase + i;
        if (e >= n_edges) return;
        const int r = eidx[e];
        const int c = eidx[n_edges + e];

        v4f va = *(const v4f*)(ab + (size_t)r * H2 + o);        // A half
        v4f vb = *(const v4f*)(ab + (size_t)c * H2 + H + o);    // B half

        float s = 0.f;
        #pragma unroll
        for (int k = 0; k < 4; ++k) {
            float t = va[k] + vb[k] + vb1[k];
            t = (t >= 0.f) ? t : NEG_SLOPE * t;                 // LeakyReLU
            s = fmaf(t, vw2[k], s);
        }
        // wave32 tree reduction
        #pragma unroll
        for (int off = 16; off > 0; off >>= 1)
            s += __shfl_down(s, off, 32);
        if (lane == 0)
            out[e] = s + bias2;
    }
}

// ---------------------------------------------------------------------------
extern "C" void kernel_launch(void* const* d_in, const int* in_sizes, int n_in,
                              void* d_out, int out_size, void* d_ws, size_t ws_size,
                              hipStream_t stream) {
    const float* z    = (const float*)d_in[0];   // [n_nodes, 128] f32
    const int*   eidx = (const int*)  d_in[1];   // [2, n_edges] int
    const float* w1   = (const float*)d_in[2];   // [128, 256] f32
    const float* b1   = (const float*)d_in[3];   // [128] f32
    const float* w2   = (const float*)d_in[4];   // [1, 128] f32
    const float* b2   = (const float*)d_in[5];   // [1] f32
    float* out        = (float*)d_out;           // [n_edges] f32

    const int n_nodes = in_sizes[0] / H;
    const int n_edges = out_size;
    const int n_pad   = (n_nodes + 15) & ~15;    // pad rows to WMMA M-tile

    // Workspace layout (offsets 256B-aligned by construction):
    //   Wp : bf16 [256 x 128]          = 64 KB
    //   zb : bf16 [n_pad x 128]        = ~25.6 MB
    //   AB : f32  [n_pad x 256]        = ~102.4 MB
    char* ws = (char*)d_ws;
    __bf16* wp = (__bf16*)ws;
    __bf16* zb = (__bf16*)(ws + 65536);
    float*  ab = (float*) (ws + 65536 + (size_t)n_pad * H * sizeof(__bf16));

    // Stage 0: pack operands to bf16 (weights transposed for contiguous B frags)
    pack_w_kernel<<<H2, H, 0, stream>>>(w1, wp);
    const int total_real = n_nodes * H;
    const int total_pad  = n_pad * H;
    pack_z_kernel<<<(total_pad / 4 + 255) / 256, 256, 0, stream>>>(
        z, zb, total_real, total_pad);

    // Stage 1: node-level fused GEMM via v_wmma_f32_16x16x32_bf16
    dim3 ggrid(n_pad / 16, 2);
    gemm_node_kernel<<<ggrid, 256, 0, stream>>>(zb, wp, ab);

    // Stage 2: per-edge gather + LeakyReLU + W2 dot (L2-resident AB)
    const int edges_per_block = 8 /*waves*/ * 8 /*EPW*/;
    const int eblocks = (n_edges + edges_per_block - 1) / edges_per_block;
    edge_kernel<<<eblocks, 256, 0, stream>>>(ab, eidx, b1, w2, b2, out, n_edges);
}
